// TiedAxialSelfAttention2d_11295763988983
// MI455X (gfx1250) — matrix operand order repaired and checked by reading, not compiled, not run on hardware
//
#include <hip/hip_runtime.h>
#include <hip/hip_bf16.h>

// ---------------------------------------------------------------------------
// Types
// ---------------------------------------------------------------------------
typedef __bf16 bf16;
typedef __attribute__((ext_vector_type(16))) __bf16 v16bf;
typedef __attribute__((ext_vector_type(8)))  __bf16 v8bf;
typedef __attribute__((ext_vector_type(8)))  float  v8f;
typedef __attribute__((ext_vector_type(4)))  unsigned int v4u;
typedef __attribute__((ext_vector_type(8)))  int v8i;
typedef __attribute__((ext_vector_type(4)))  int v4i;

#define EMBED 768
#define S_DIM 64
#define SL    49152          // S_DIM * EMBED
#define NHEAD 12
#define DHEAD 64

// ---------------------------------------------------------------------------
// Tensor Data Mover availability (device pass only; host pass falls back)
// ---------------------------------------------------------------------------
#if defined(__has_builtin)
#if __has_builtin(__builtin_amdgcn_tensor_load_to_lds) && \
    __has_builtin(__builtin_amdgcn_s_wait_tensorcnt)
#define HAVE_TDM 1
#endif
#endif
#ifndef HAVE_TDM
#define HAVE_TDM 0
#endif

// low 32 bits of a flat LDS pointer == byte offset in the wave's LDS space
__device__ __forceinline__ unsigned lds_off(const void* p) {
  return (unsigned)(unsigned long long)p;
}

#if HAVE_TDM
// Issue a 2D TDM tile load: dim0_elems contiguous bf16 per line, dim1_lines
// lines separated by stride0_elems (bf16 units). LDS dest packed
// [dim1][dim0]. Tensor dims == tile dims (no OOB clipping needed; all our
// tiles are in-bounds by construction).
__device__ __forceinline__ void tdm_load_tile_2d(const void* gaddr,
                                                 unsigned lds_byte_off,
                                                 int dim0_elems, int dim1_lines,
                                                 long long stride0_elems) {
  const unsigned long long ga = (unsigned long long)gaddr;
  const unsigned long long s0 = (unsigned long long)stride0_elems;
  const unsigned td0 = (unsigned)dim0_elems;
  const unsigned td1 = (unsigned)dim1_lines;
  v4u g0;
  g0[0] = 1u;                                    // count=1 (valid user D#)
  g0[1] = lds_byte_off;                          // lds_addr (bytes)
  g0[2] = (unsigned)ga;                          // global_addr[31:0]
  g0[3] = (unsigned)((ga >> 32) & 0x01ffffffu)   // global_addr[56:32]
          | (2u << 30);                          // type = 2 ("image")
  v8i g1;
  g1[0] = (int)(1u << 16);                       // data_size=1 (2B), mask=0
  g1[1] = (int)((td0 & 0xffffu) << 16);          // tensor_dim0[15:0] @ [63:48]
  g1[2] = (int)((td0 >> 16) | ((td1 & 0xffffu) << 16)); // td0 hi | td1 lo
  g1[3] = (int)((td1 >> 16) | (td0 << 16));      // td1 hi | tile_dim0 @[127:112]
  g1[4] = (int)(td1 & 0xffffu);                  // tile_dim1; tile_dim2=0 (2D)
  g1[5] = (int)(unsigned)s0;                     // dim0_stride[31:0]
  g1[6] = (int)((unsigned)(s0 >> 32) & 0xffffu); // dim0_stride[47:32]
  g1[7] = 0;                                     // dim1_stride unused (2D)
  v4i z4 = {};
#if __clang_major__ >= 23
  v8i z8 = {};
  __builtin_amdgcn_tensor_load_to_lds(g0, g1, z4, z4, z8, 0);
#else
  __builtin_amdgcn_tensor_load_to_lds(g0, g1, z4, z4, 0);
#endif
}
#define TDM_WAIT() __builtin_amdgcn_s_wait_tensorcnt(0)
#else
__device__ __forceinline__ void tdm_load_tile_2d(const void*, unsigned, int,
                                                 int, long long) {}
#define TDM_WAIT()
#endif

// ---------------------------------------------------------------------------
// WMMA helpers (CDNA5 wave32, 16x16x32 bf16 -> f32)
// ---------------------------------------------------------------------------
__device__ __forceinline__ v8f wmma_bf16(v16bf a, v16bf b, v8f c) {
  return __builtin_amdgcn_wmma_f32_16x16x32_bf16(false, a, false, b,
                                                 (short)0, c, false, false);
}

// A fragment (16x32 bf16) from LDS tile S[m][k], row stride ld.
__device__ __forceinline__ v16bf frag_a(const bf16* S, int ld, int m0, int k0,
                                        int lane) {
  const int half = lane >> 4;
  const bf16* p = S + (m0 + (lane & 15)) * ld + k0;
  v8bf lo = *(const v8bf*)(p + half * 8);
  v8bf hi = *(const v8bf*)(p + 16 + half * 8);
  v16bf r;
#pragma unroll
  for (int i = 0; i < 8; ++i) { r[i] = lo[i]; r[i + 8] = hi[i]; }
  return r;
}

// B fragment (32x16 bf16) from LDS tile stored TRANSPOSED: S[n][k], stride ld.
__device__ __forceinline__ v16bf frag_b(const bf16* S, int ld, int n0, int k0,
                                        int lane) {
  const int half = lane >> 4;
  const bf16* p = S + (n0 + (lane & 15)) * ld + k0;
  v8bf lo = *(const v8bf*)(p + half * 8);
  v8bf hi = *(const v8bf*)(p + 16 + half * 8);
  v16bf r;
#pragma unroll
  for (int i = 0; i < 8; ++i) { r[i] = lo[i]; r[i + 8] = hi[i]; }
  return r;
}

// ---------------------------------------------------------------------------
// Generic batched bf16 WMMA GEMM: C[m][n] = sum_k A[m][k]*B[k][n] (+bias[m])
//   TA: A stored [K][M] (ld=M) ; else [M][K] (ld=K)
//   TB: B stored [N][K] (ld=K) ; else [K][N] (ld=N)
// Contiguous-staging operands (!TA for A, TB for B) go through the TDM.
// ---------------------------------------------------------------------------
#define BM 128
#define BN 128
#define BK 32
#define BKP 40   // padded LDS K-stride for manually staged tiles

template <bool TA, bool TB, typename OutT>
__global__ __launch_bounds__(256) void gemm_wmma(
    const bf16* __restrict__ Ag, const bf16* __restrict__ Bg,
    OutT* __restrict__ Cg, const float* __restrict__ bias,
    int M, int N, int K, long long sA, long long sB, long long sC) {
  constexpr bool kTdmA = HAVE_TDM && !TA;
  constexpr bool kTdmB = HAVE_TDM && TB;
  constexpr int LDSA = kTdmA ? BK : BKP;
  constexpr int LDSB = kTdmB ? BK : BKP;

  __shared__ bf16 As[BM * BKP];
  __shared__ bf16 Bs[BN * BKP];

  const int tid  = threadIdx.x;
  const int lane = tid & 31;
  const int wave = tid >> 5;
  const int wrow = wave >> 2;  // 0..1 -> 64 rows
  const int wcol = wave & 3;   // 0..3 -> 32 cols
  const int bM = blockIdx.y * BM;
  const int bN = blockIdx.x * BN;
  const bf16* A = Ag + (long long)blockIdx.z * sA;
  const bf16* B = Bg + (long long)blockIdx.z * sB;
  OutT* C = Cg + (long long)blockIdx.z * sC;
  const int lda = TA ? M : K;
  const int ldb = TB ? K : N;

  v8f acc[4][2];
  const v8f zero = {};
#pragma unroll
  for (int i = 0; i < 4; ++i)
#pragma unroll
    for (int j = 0; j < 2; ++j) acc[i][j] = zero;

  for (int k0 = 0; k0 < K; k0 += BK) {
    // ---- TDM staging (wave 0 issues; branch keeps other waves from issuing)
    if constexpr (kTdmA || kTdmB) {
      if (tid < 32) {
        if constexpr (kTdmA)
          tdm_load_tile_2d(A + (long long)bM * lda + k0, lds_off(As), BK, BM,
                           lda);
        if constexpr (kTdmB)
          tdm_load_tile_2d(B + (long long)bN * ldb + k0, lds_off(Bs), BK, BN,
                           ldb);
        TDM_WAIT();
      }
    }
    // ---- manual staging for operands the TDM cannot handle ----
    if constexpr (!kTdmA) {
      if (!TA) {
        const int r  = tid >> 1;
        const int cs = (tid & 1) * 16;
        const bf16* gp = A + (long long)(bM + r) * lda + (k0 + cs);
        v8bf a0 = *(const v8bf*)gp;
        v8bf a1 = *(const v8bf*)(gp + 8);
        *(v8bf*)&As[r * LDSA + cs]     = a0;
        *(v8bf*)&As[r * LDSA + cs + 8] = a1;
      } else {
        const int kk = tid >> 3;
        const int ms = (tid & 7) * 16;
        const bf16* gp = A + (long long)(k0 + kk) * lda + (bM + ms);
        v8bf a0 = *(const v8bf*)gp;
        v8bf a1 = *(const v8bf*)(gp + 8);
#pragma unroll
        for (int i = 0; i < 8; ++i) {
          As[(ms + i) * LDSA + kk]     = a0[i];
          As[(ms + 8 + i) * LDSA + kk] = a1[i];
        }
      }
    }
    if constexpr (!kTdmB) {
      if (TB) {
        const int r  = tid >> 1;
        const int cs = (tid & 1) * 16;
        const bf16* gp = B + (long long)(bN + r) * ldb + (k0 + cs);
        v8bf b0 = *(const v8bf*)gp;
        v8bf b1 = *(const v8bf*)(gp + 8);
        *(v8bf*)&Bs[r * LDSB + cs]     = b0;
        *(v8bf*)&Bs[r * LDSB + cs + 8] = b1;
      } else {
        const int kk = tid >> 3;
        const int ns = (tid & 7) * 16;
        const bf16* gp = B + (long long)(k0 + kk) * ldb + (bN + ns);
        v8bf b0 = *(const v8bf*)gp;
        v8bf b1 = *(const v8bf*)(gp + 8);
#pragma unroll
        for (int i = 0; i < 8; ++i) {
          Bs[(ns + i) * LDSB + kk]     = b0[i];
          Bs[(ns + 8 + i) * LDSB + kk] = b1[i];
        }
      }
    }
    __syncthreads();

    // prefetch next K-tile (gfx1250 global_prefetch_b8) for manual operands
    if (k0 + BK < K) {
      const int kn = k0 + BK;
      if constexpr (!kTdmA)
        __builtin_prefetch(A + (long long)(TA ? kn : bM) * lda + (TA ? bM : kn),
                           0, 1);
      if constexpr (!kTdmB)
        __builtin_prefetch(B + (long long)(TB ? bN : kn) * ldb + (TB ? kn : bN),
                           0, 1);
    }

    // ---- compute: 8 WMMAs / wave / K-step ----
    v16bf af[4];
#pragma unroll
    for (int mt = 0; mt < 4; ++mt)
      af[mt] = frag_a(As, LDSA, wrow * 64 + mt * 16, 0, lane);
#pragma unroll
    for (int nt = 0; nt < 2; ++nt) {
      v16bf bfr = frag_b(Bs, LDSB, wcol * 32 + nt * 16, 0, lane);
#pragma unroll
      for (int mt = 0; mt < 4; ++mt)
        acc[mt][nt] = wmma_bf16(af[mt], bfr, acc[mt][nt]);
    }
    __syncthreads();
  }

  // ---- epilogue: C layout VGPR r -> (m0+8*half+r, n0+(lane&15)) ----
  const int half = lane >> 4;
  const int lcol = lane & 15;
#pragma unroll
  for (int mt = 0; mt < 4; ++mt) {
#pragma unroll
    for (int nt = 0; nt < 2; ++nt) {
      const int col = bN + wcol * 32 + nt * 16 + lcol;
#pragma unroll
      for (int r = 0; r < 8; ++r) {
        const int row = bM + wrow * 64 + mt * 16 + half * 8 + r;
        float v = acc[mt][nt][r];
        if (bias) v += bias[row];
        C[(long long)row * N + col] = (OutT)v;
      }
    }
  }
}

// ---------------------------------------------------------------------------
// fp32 -> bf16 conversion (grid-stride)
// ---------------------------------------------------------------------------
__global__ void cvt_f32_bf16(const float* __restrict__ in, bf16* __restrict__ out,
                             long long n) {
  long long i = (long long)blockIdx.x * 256 + threadIdx.x;
  const long long stride = (long long)gridDim.x * 256;
  for (; i < n; i += stride) out[i] = (bf16)in[i];
}

// ---------------------------------------------------------------------------
// Row softmax over last axis (len), fp32 in -> bf16 out
// ---------------------------------------------------------------------------
__global__ __launch_bounds__(256) void softmax_rows(
    const float* __restrict__ in, bf16* __restrict__ out, int len) {
  __shared__ float red[256];
  const long long base = (long long)blockIdx.x * len;
  const int tid = threadIdx.x;
  float m = -3.0e38f;
  for (int j = tid; j < len; j += 256) m = fmaxf(m, in[base + j]);
  red[tid] = m;
  __syncthreads();
  for (int s = 128; s > 0; s >>= 1) {
    if (tid < s) red[tid] = fmaxf(red[tid], red[tid + s]);
    __syncthreads();
  }
  m = red[0];
  __syncthreads();
  float sum = 0.f;
  for (int j = tid; j < len; j += 256) sum += __expf(in[base + j] - m);
  red[tid] = sum;
  __syncthreads();
  for (int s = 128; s > 0; s >>= 1) {
    if (tid < s) red[tid] += red[tid + s];
    __syncthreads();
  }
  const float inv = 1.0f / red[0];
  for (int j = tid; j < len; j += 256)
    out[base + j] = (bf16)(__expf(in[base + j] - m) * inv);
}

// ---------------------------------------------------------------------------
// out = LayerNorm_L(xa + xb) * g + b   (L = 768, one block per row)
// ---------------------------------------------------------------------------
__global__ __launch_bounds__(256) void add_ln(
    const float* __restrict__ xa, const float* __restrict__ xb,
    const float* __restrict__ g, const float* __restrict__ b,
    float* __restrict__ out, bf16* __restrict__ outb) {
  const int L = EMBED;
  const long long base = (long long)blockIdx.x * L;
  __shared__ float r1[256], r2[256];
  const int tid = threadIdx.x;
  float v[3], s = 0.f, s2 = 0.f;
#pragma unroll
  for (int i = 0; i < 3; ++i) {
    const int j = tid + i * 256;
    const float t = xa[base + j] + xb[base + j];
    v[i] = t; s += t; s2 += t * t;
  }
  r1[tid] = s; r2[tid] = s2;
  __syncthreads();
  for (int st = 128; st > 0; st >>= 1) {
    if (tid < st) { r1[tid] += r1[tid + st]; r2[tid] += r2[tid + st]; }
    __syncthreads();
  }
  const float mean = r1[0] / L;
  const float var  = r2[0] / L - mean * mean;
  const float inv  = rsqrtf(var + 1e-5f);
#pragma unroll
  for (int i = 0; i < 3; ++i) {
    const int j = tid + i * 256;
    const float o = (v[i] - mean) * inv * g[j] + b[j];
    out[base + j] = o;
    if (outb) outb[base + j] = (bf16)o;
  }
}

// ---------------------------------------------------------------------------
// Transpose bf16 [2304][64][768] -> [2304][768][64]
// ---------------------------------------------------------------------------
__global__ __launch_bounds__(256) void transpose_sl(
    const bf16* __restrict__ in, bf16* __restrict__ out) {
  const int o  = blockIdx.y;
  const int l0 = blockIdx.x * 64;
  __shared__ bf16 t[64][72];
  {
    const int s  = threadIdx.x >> 2;
    const int ls = (threadIdx.x & 3) * 16;
    const bf16* gp = in + (long long)o * SL + (long long)s * EMBED + l0 + ls;
    v8bf a = *(const v8bf*)gp;
    v8bf b = *(const v8bf*)(gp + 8);
    *(v8bf*)&t[s][ls]     = a;
    *(v8bf*)&t[s][ls + 8] = b;
  }
  __syncthreads();
  {
    const int li = threadIdx.x >> 2;
    const int ss = (threadIdx.x & 3) * 16;
    v8bf o0, o1;
#pragma unroll
    for (int i = 0; i < 8; ++i) { o0[i] = t[ss + i][li]; o1[i] = t[ss + 8 + i][li]; }
    bf16* op = out + (long long)o * SL + (long long)(l0 + li) * 64 + ss;
    *(v8bf*)op       = o0;
    *(v8bf*)(op + 8) = o1;
  }
}

// ---------------------------------------------------------------------------
// Fused column attention: one block per (l, h), 128 threads = 4 waves.
// Yt layout: [o=2304][l=768][s=64] bf16.
// ---------------------------------------------------------------------------
#if HAVE_TDM
#define LDV 64   // V tile staged by TDM, packed stride
#else
#define LDV 72
#endif

__global__ __launch_bounds__(128) void col_attn_kernel(
    const bf16* __restrict__ Yt, float* __restrict__ outp) {
  const int l   = blockIdx.x;
  const int h   = blockIdx.y;
  const int tid = threadIdx.x;
  const int lane = tid & 31;
  const int wave = tid >> 5;  // 0..3
  const int LD = 72;

  __shared__ bf16 Qs[64 * 72];   // [i=s][c]   (A for logits)
  __shared__ bf16 Ks[64 * 72];   // [j][c]     (B^T for logits)
  __shared__ bf16 Vs[64 * 72];   // [c'][j]    (A for apply)
  __shared__ bf16 Pa[64 * 72];   // attn [i][j] (B^T for apply)
  __shared__ float Lg[64 * 66];  // logits fp32

  const long long bq = (long long)(h * 64) * SL + (long long)l * 64;
  const long long bk = (long long)(EMBED + h * 64) * SL + (long long)l * 64;
  const long long bv = (long long)(2 * EMBED + h * 64) * SL + (long long)l * 64;

#if HAVE_TDM
  // V tile (64 rows of 64 contiguous bf16, row stride SL) via the TDM
  if (tid < 32) {
    tdm_load_tile_2d(Yt + bv, lds_off(Vs), 64, 64, (long long)SL);
  }
#endif

  // ---- stage Q^T, K^T (and V without TDM) ----
  {
    const int c  = tid >> 1;
    const int s0 = (tid & 1) * 32;
    const bf16* q = Yt + bq + (long long)c * SL + s0;
    const bf16* k = Yt + bk + (long long)c * SL + s0;
#if !HAVE_TDM
    const bf16* v = Yt + bv + (long long)c * SL + s0;
#endif
#pragma unroll
    for (int b = 0; b < 4; ++b) {
      v8bf qv = *(const v8bf*)(q + b * 8);
      v8bf kv = *(const v8bf*)(k + b * 8);
#pragma unroll
      for (int i = 0; i < 8; ++i) {
        const int s = s0 + b * 8 + i;
        Qs[s * LD + c] = qv[i];
        Ks[s * LD + c] = kv[i];
      }
#if !HAVE_TDM
      v8bf vv = *(const v8bf*)(v + b * 8);
      *(v8bf*)&Vs[c * LDV + s0 + b * 8] = vv;
#endif
    }
  }
#if HAVE_TDM
  if (tid < 32) TDM_WAIT();
#endif
  __syncthreads();

  // ---- logits: wave owns rows [wave*16, wave*16+16) ----
  {
    v8f acc[4];
    const v8f zero = {};
#pragma unroll
    for (int nt = 0; nt < 4; ++nt) acc[nt] = zero;
#pragma unroll
    for (int kk = 0; kk < 64; kk += 32) {
      v16bf a = frag_a(Qs, LD, wave * 16, kk, lane);
#pragma unroll
      for (int nt = 0; nt < 4; ++nt) {
        v16bf b = frag_b(Ks, LD, nt * 16, kk, lane);
        acc[nt] = wmma_bf16(a, b, acc[nt]);
      }
    }
    const int half = lane >> 4, lc = lane & 15;
#pragma unroll
    for (int nt = 0; nt < 4; ++nt)
#pragma unroll
      for (int r = 0; r < 8; ++r)
        Lg[(wave * 16 + half * 8 + r) * 66 + nt * 16 + lc] = acc[nt][r];
  }
  __syncthreads();

  // ---- softmax over j (one thread per row i) ----
  if (tid < 64) {
    float m = -3.0e38f;
    for (int j = 0; j < 64; ++j) m = fmaxf(m, Lg[tid * 66 + j]);
    float s = 0.f;
    for (int j = 0; j < 64; ++j) {
      const float e = __expf(Lg[tid * 66 + j] - m);
      Lg[tid * 66 + j] = e;
      s += e;
    }
    const float inv = 1.0f / s;
    for (int j = 0; j < 64; ++j)
      Pa[tid * LD + j] = (bf16)(Lg[tid * 66 + j] * inv);
  }
  __syncthreads();

  // ---- apply: out(c',i) = sum_j V[c'][j] * attn[i][j] ----
  {
    v8f acc[4];
    const v8f zero = {};
#pragma unroll
    for (int nt = 0; nt < 4; ++nt) acc[nt] = zero;
#pragma unroll
    for (int kk = 0; kk < 64; kk += 32) {
      v16bf a = frag_a(Vs, LDV, wave * 16, kk, lane);
#pragma unroll
      for (int nt = 0; nt < 4; ++nt) {
        v16bf b = frag_b(Pa, LD, nt * 16, kk, lane);
        acc[nt] = wmma_bf16(a, b, acc[nt]);
      }
    }
    const int half = lane >> 4, lc = lane & 15;
#pragma unroll
    for (int nt = 0; nt < 4; ++nt) {
#pragma unroll
      for (int r = 0; r < 8; ++r) {
        const int cp = wave * 16 + half * 8 + r;
        const int i  = nt * 16 + lc;
        outp[(long long)(h * 64 + cp) * SL + (long long)i * EMBED + l] =
            acc[nt][r];
      }
    }
  }
}

// ---------------------------------------------------------------------------
// Host-side orchestration
// ---------------------------------------------------------------------------
extern "C" void kernel_launch(void* const* d_in, const int* in_sizes, int n_in,
                              void* d_out, int out_size, void* d_ws,
                              size_t ws_size, hipStream_t stream) {
  (void)in_sizes; (void)n_in; (void)out_size; (void)ws_size;

  const float* x     = (const float*)d_in[0];
  const float* w_row = (const float*)d_in[1];
  const float* b_row = (const float*)d_in[2];
  const float* w_col = (const float*)d_in[3];
  const float* b_col = (const float*)d_in[4];
  const float* g1    = (const float*)d_in[5];
  const float* be1   = (const float*)d_in[6];
  const float* g2    = (const float*)d_in[7];
  const float* be2   = (const float*)d_in[8];
  float* outf = (float*)d_out;

  const size_t NE_X = (size_t)EMBED * SL;
  const size_t NE_W = (size_t)3 * EMBED * EMBED;
  const size_t NE_Y = (size_t)3 * EMBED * SL;
  const size_t NE_A = (size_t)NHEAD * EMBED * EMBED;

  char* w = (char*)d_ws;
  size_t off = 0;
  auto take = [&](size_t bytes) -> void* {
    void* p = w + off;
    off += (bytes + 255) & ~size_t(255);
    return p;
  };
  bf16*  xbf    = (bf16*)take(NE_X * 2);
  bf16*  wrbf   = (bf16*)take(NE_W * 2);
  bf16*  wcbf   = (bf16*)take(NE_W * 2);
  bf16*  Y      = (bf16*)take(NE_Y * 2);   // QKV (row pass, reused for col)
  float* logits = (float*)take(NE_A * 4);
  bf16*  attn   = (bf16*)take(NE_A * 2);
  float* rowout = (float*)take(NE_X * 4);  // reused as colout
  float* out1   = (float*)take(NE_X * 4);
  bf16*  Yt     = (bf16*)take(NE_Y * 2);   // transposed col QKV [o][l][s]
  bf16*  out1bf = xbf;                     // xbf dead after row QKV GEMM

  const long long HS = 64LL * SL;
  const long long AS = (long long)EMBED * EMBED;

  // 1) fp32 -> bf16
  cvt_f32_bf16<<<2048, 256, 0, stream>>>(x, xbf, (long long)NE_X);
  cvt_f32_bf16<<<512, 256, 0, stream>>>(w_row, wrbf, (long long)NE_W);
  cvt_f32_bf16<<<512, 256, 0, stream>>>(w_col, wcbf, (long long)NE_W);

  // 2) QKV row: Y = Wr @ X + b_row
  gemm_wmma<false, false, bf16><<<dim3(SL / BN, 2304 / BM, 1), 256, 0, stream>>>(
      wrbf, xbf, Y, b_row, 2304, SL, EMBED, 0, 0, 0);

  // 3) Row logits (per head): L[i][j] = sum_cs Q[cs][i] K[cs][j]  (transA)
  gemm_wmma<true, false, float><<<dim3(EMBED / BN, EMBED / BM, NHEAD), 256, 0,
                                  stream>>>(
      Y, Y + (long long)EMBED * SL, logits, nullptr, EMBED, EMBED, 4096, HS, HS,
      AS);

  // 4) Softmax over j -> bf16 attn
  softmax_rows<<<NHEAD * EMBED, 256, 0, stream>>>(logits, attn, EMBED);

  // 5) Row apply (per head): O[cs][i] = sum_j V[cs][j] attn[i][j]  (transB)
  gemm_wmma<false, true, float><<<dim3(EMBED / BN, 4096 / BM, NHEAD), 256, 0,
                                  stream>>>(
      Y + 2LL * EMBED * SL, attn, rowout, nullptr, 4096, EMBED, EMBED, HS, AS,
      HS);

  // 6) out1 = LN(x + rowout); bf16 shadow
  add_ln<<<SL, 256, 0, stream>>>(x, rowout, g1, be1, out1, out1bf);

  // 7) QKV col: Y = Wc @ out1 + b_col
  gemm_wmma<false, false, bf16><<<dim3(SL / BN, 2304 / BM, 1), 256, 0, stream>>>(
      wcbf, out1bf, Y, b_col, 2304, SL, EMBED, 0, 0, 0);

  // 8) Transpose to [o][l][s]
  transpose_sl<<<dim3(EMBED / 64, 2304), 256, 0, stream>>>(Y, Yt);

  // 9) Fused column attention -> colout
  col_attn_kernel<<<dim3(EMBED, NHEAD), 128, 0, stream>>>(Yt, rowout);

  // 10) final: out = LN(out1 + colout)
  add_ln<<<SL, 256, 0, stream>>>(out1, rowout, g2, be2, outf, nullptr);
}
